// DecoderLayer_2671469658570
// MI455X (gfx1250) — matrix operand order repaired and checked by reading, not compiled, not run on hardware
//
#include <hip/hip_runtime.h>
#include <math.h>

typedef __attribute__((ext_vector_type(16))) __bf16 v16bf;
typedef __attribute__((ext_vector_type(8)))  __bf16 v8bf;
typedef __attribute__((ext_vector_type(8)))  float  v8f;

// ---------- small helpers ----------
static __device__ __forceinline__ unsigned short f2bf(float f) {
    union { float f; unsigned u; } c; c.f = f;
    unsigned u = c.u;
    u += 0x7fffu + ((u >> 16) & 1u);          // round-to-nearest-even
    return (unsigned short)(u >> 16);
}

// CDNA5 async global->LDS copy (16 bytes per lane), tracked by ASYNCcnt.
// LDS destination address = low 32 bits of the generic pointer (AS3 offset).
static __device__ __forceinline__ void async_copy16(void* lds, const void* g) {
    unsigned l = (unsigned)(unsigned long long)lds;
    unsigned long long ga = (unsigned long long)g;
    asm volatile("global_load_async_to_lds_b128 %0, %1, off"
                 :: "v"(l), "v"(ga) : "memory");
}
static __device__ __forceinline__ void wait_async0() {
    asm volatile("s_wait_asynccnt 0x0" ::: "memory");
}

// Load a 16x32 bf16 A-operand fragment for this lane.
// Layout (16-bit A, 16x32): lane m = lane&15, half = lane>>4.
// element e -> K = half*16 + e: one contiguous 16-element run
// at offset half*16 within the 32-wide K window (matches B operand reads).
static __device__ __forceinline__ v16bf load_afrag(const unsigned short* rowPtr, int half) {
    v8bf lo = *(const v8bf*)(rowPtr + half * 16);
    v8bf hi = *(const v8bf*)(rowPtr + half * 16 + 8);
    v16bf a;
#pragma unroll
    for (int i = 0; i < 8; ++i) { a[i] = lo[i]; a[8 + i] = hi[i]; }
    return a;
}

// ---------- f32 -> bf16 convert ----------
__global__ void cvt_kernel(const float* __restrict__ in, unsigned short* __restrict__ out, int n) {
    int i = blockIdx.x * blockDim.x + threadIdx.x;
    if (i < n) out[i] = f2bf(in[i]);
}

// ---------- LayerNorm (row = 1024), f32 in -> bf16 out ----------
__global__ __launch_bounds__(256)
void ln_kernel(const float* __restrict__ x, const float* __restrict__ g,
               const float* __restrict__ bta, unsigned short* __restrict__ out) {
    __shared__ float red[8];
    __shared__ float bcast;
    const int row = blockIdx.x;
    const int tid = threadIdx.x;
    const float* xr = x + (size_t)row * 1024;
    float v[4];
#pragma unroll
    for (int i = 0; i < 4; ++i) v[i] = xr[tid + i * 256];
    float s = v[0] + v[1] + v[2] + v[3];
#pragma unroll
    for (int off = 16; off > 0; off >>= 1) s += __shfl_xor(s, off, 32);
    if ((tid & 31) == 0) red[tid >> 5] = s;
    __syncthreads();
    if (tid == 0) {
        float t = 0.f;
        for (int i = 0; i < 8; ++i) t += red[i];
        bcast = t * (1.0f / 1024.0f);
    }
    __syncthreads();
    float mu = bcast;
    float d2 = 0.f;
#pragma unroll
    for (int i = 0; i < 4; ++i) { float d = v[i] - mu; d2 += d * d; }
#pragma unroll
    for (int off = 16; off > 0; off >>= 1) d2 += __shfl_xor(d2, off, 32);
    __syncthreads();
    if ((tid & 31) == 0) red[tid >> 5] = d2;
    __syncthreads();
    if (tid == 0) {
        float t = 0.f;
        for (int i = 0; i < 8; ++i) t += red[i];
        bcast = t * (1.0f / 1024.0f);
    }
    __syncthreads();
    float rs = rsqrtf(bcast + 1e-5f);
#pragma unroll
    for (int i = 0; i < 4; ++i) {
        int c = tid + i * 256;
        out[(size_t)row * 1024 + c] = f2bf((v[i] - mu) * rs * g[c] + bta[c]);
    }
}

// ---------- bf16 GEMM: C[M,N] = act(A[M,K] * W[K,N] + bias) (+ residual) ----------
// 128x128 block tile, 256 threads (8 waves), wave tile 32x64 (2x4 WMMA tiles), K-step 32.
// Double-buffered LDS. A tile: async global->LDS. W tile: global loads issued BEFORE the
// WMMA burst, transpose ds-store scatter committed AFTER it (loadcnt stall hidden by math).
__global__ __launch_bounds__(256)
void gemm_bf16_kernel(const unsigned short* __restrict__ A,
                      const unsigned short* __restrict__ W,
                      const float* __restrict__ bias,
                      const float* __restrict__ residual,
                      float* __restrict__ outF,
                      unsigned short* __restrict__ outB,
                      int M, int N, int K, int doGelu) {
    __shared__ unsigned short Atile[2][128 * 32];   // [m][k]
    __shared__ unsigned short Wt[2][128 * 32];      // [n][k] (transposed: B-operand contiguous)

    const int tid  = threadIdx.x;
    const int wave = tid >> 5;
    const int lane = tid & 31;
    const int l15  = lane & 15;
    const int half = lane >> 4;

    const int mBase = blockIdx.y * 128;
    const int nBase = blockIdx.x * 128;
    const int waveM = wave >> 1;   // 0..3
    const int waveN = wave & 1;    // 0..1

    v8f acc[2][4];
#pragma unroll
    for (int i = 0; i < 2; ++i)
#pragma unroll
        for (int j = 0; j < 4; ++j) acc[i][j] = {};

    // loader roles
    const int rowA  = tid >> 1;   // 0..127
    const int halfA = tid & 1;    // 16 elems each
    const int kk    = tid >> 3;   // 0..31
    const int ng    = tid & 7;    // 0..7 -> 16 cols each

    const unsigned short* aSrc = A + (size_t)(mBase + rowA) * K + halfA * 16;
    unsigned short* aDstBase0 = &Atile[0][rowA * 32 + halfA * 16];
    unsigned short* aDstBase1 = &Atile[1][rowA * 32 + halfA * 16];

    uint4 wu0, wu1;   // in-flight W tile (held across the WMMA burst)

    auto stageA = [&](int k0, int buf) {
        unsigned short* d = buf ? aDstBase1 : aDstBase0;
        async_copy16(d,     aSrc + k0);
        async_copy16(d + 8, aSrc + k0 + 8);
    };
    auto loadW = [&](int k0) {
        const uint4* src = (const uint4*)(W + (size_t)(k0 + kk) * N + nBase + ng * 16);
        wu0 = src[0]; wu1 = src[1];
    };
    auto commitW = [&](int buf) {
        union { uint4 v[2]; unsigned short s[16]; } u;
        u.v[0] = wu0; u.v[1] = wu1;
        unsigned short* wt = &Wt[buf][0];
#pragma unroll
        for (int i = 0; i < 16; ++i) wt[(ng * 16 + i) * 32 + kk] = u.s[i];
    };

    const int nIter = K >> 5;
    stageA(0, 0);
    loadW(0);
    commitW(0);
    wait_async0();
    __syncthreads();

    for (int it = 0; it < nIter; ++it) {
        const int cur  = it & 1;
        const bool more = (it + 1 < nIter);
        if (more) {                          // issue next tile's loads before computing
            stageA((it + 1) * 32, cur ^ 1);
            loadW((it + 1) * 32);
        }

        v16bf afr[2], bfr[4];
#pragma unroll
        for (int mt = 0; mt < 2; ++mt) {
            int m = waveM * 32 + mt * 16 + l15;
            afr[mt] = load_afrag(&Atile[cur][m * 32], half);
        }
#pragma unroll
        for (int nt = 0; nt < 4; ++nt) {
            int n = waveN * 64 + nt * 16 + l15;
            bfr[nt] = *(const v16bf*)(&Wt[cur][n * 32 + half * 16]);
        }
#pragma unroll
        for (int nt = 0; nt < 4; ++nt)
#pragma unroll
            for (int mt = 0; mt < 2; ++mt)
                acc[mt][nt] = __builtin_amdgcn_wmma_f32_16x16x32_bf16(
                    false, afr[mt], false, bfr[nt], (short)0, acc[mt][nt], false, false);

        if (more) commitW(cur ^ 1);          // loadcnt stall now lands after 8 WMMAs
        wait_async0();
        __syncthreads();
    }

    // epilogue: C layout -> lane holds column n = l15, VGPR r -> row m = r + 8*half
#pragma unroll
    for (int mt = 0; mt < 2; ++mt) {
#pragma unroll
        for (int nt = 0; nt < 4; ++nt) {
            int col = nBase + waveN * 64 + nt * 16 + l15;
            float bv = bias ? bias[col] : 0.0f;
#pragma unroll
            for (int r = 0; r < 8; ++r) {
                int row = mBase + waveM * 32 + mt * 16 + r + 8 * half;
                float v = acc[mt][nt][r] + bv;
                if (doGelu) v = 0.5f * v * (1.0f + erff(v * 0.70710678118f));
                if (residual) v += residual[(size_t)row * N + col];
                if (outF) outF[(size_t)row * N + col] = v;
                if (outB) outB[(size_t)row * N + col] = f2bf(v);
            }
        }
    }
}

// ---------- flash attention: O = softmax(Q K^T / 8 [causal]) V, Dk = 64 ----------
// grid (Tq/64, B*H), block 128 (4 waves); each wave owns 16 q-rows. KV chunk = 32.
// Double-buffered K/V staging: K via async global->LDS, V loads issued before the
// WMMA/softmax phase and the transpose scatter committed after it.
__global__ __launch_bounds__(128)
void attn_kernel(const unsigned short* __restrict__ Q,
                 const unsigned short* __restrict__ Kb,
                 const unsigned short* __restrict__ Vb,
                 unsigned short* __restrict__ Ob,
                 int T, int Tk, int causal) {
    __shared__ unsigned short Kt[2][32 * 64];   // [kv][dk]  == B operand of Q*K^T
    __shared__ unsigned short Vt[2][64 * 32];   // [dk][kv]  == B operand of P*V
    __shared__ unsigned short Pt[4][16 * 32];   // per-wave P [m][kv]

    const int tid  = threadIdx.x;
    const int wave = tid >> 5;
    const int lane = tid & 31;
    const int l15  = lane & 15;
    const int half = lane >> 4;

    const int bIdx    = blockIdx.y >> 4;
    const int hIdx    = blockIdx.y & 15;
    const int qTile   = blockIdx.x;
    const int headOff = hIdx * 64;
    const int D       = 1024;

    const size_t qRowBase = (size_t)bIdx * T;
    const size_t kRowBase = (size_t)bIdx * Tk;
    const int qRow0 = qTile * 64 + wave * 16;

    // Q operand fragments (dk = 64 -> 2 K-steps), loaded once from global
    v16bf aq[2];
    {
        const unsigned short* qp = Q + (qRowBase + qRow0 + l15) * D + headOff;
        aq[0] = load_afrag(qp, half);
        aq[1] = load_afrag(qp + 32, half);
    }

    v8f accO[4];
#pragma unroll
    for (int i = 0; i < 4; ++i) accO[i] = {};
    float mrow[8], lrow[8];
#pragma unroll
    for (int r = 0; r < 8; ++r) { mrow[r] = -__builtin_inff(); lrow[r] = 0.0f; }

    const int kvEnd  = causal ? (qTile * 64 + 64) : Tk;
    const int nChunk = kvEnd >> 5;
    const int ldr    = tid >> 2;   // 0..31 kv row
    const int ldp    = tid & 3;    // 16 dk cols each

    uint4 vu0, vu1;   // in-flight V chunk

    auto stageK = [&](int c, int buf) {
        const unsigned short* ks = Kb + (kRowBase + c * 32 + ldr) * D + headOff + ldp * 16;
        unsigned short* kd = &Kt[buf][ldr * 64 + ldp * 16];
        async_copy16(kd,     ks);
        async_copy16(kd + 8, ks + 8);
    };
    auto loadV = [&](int c) {
        const uint4* vs = (const uint4*)(Vb + (kRowBase + c * 32 + ldr) * D + headOff + ldp * 16);
        vu0 = vs[0]; vu1 = vs[1];
    };
    auto commitV = [&](int buf) {
        union { uint4 v[2]; unsigned short s[16]; } u;
        u.v[0] = vu0; u.v[1] = vu1;
        unsigned short* vt = &Vt[buf][0];
#pragma unroll
        for (int i = 0; i < 16; ++i) vt[(ldp * 16 + i) * 32 + ldr] = u.s[i];
    };

    stageK(0, 0);
    loadV(0);
    commitV(0);
    wait_async0();
    __syncthreads();

    for (int c = 0; c < nChunk; ++c) {
        const int cur  = c & 1;
        const bool more = (c + 1 < nChunk);
        const int kv0  = c * 32;
        if (more) {                          // issue next chunk's loads first
            stageK(c + 1, cur ^ 1);
            loadV(c + 1);
        }

        // S = Q K^T over 2 n-tiles of 16 kv
        v8f s[2];
#pragma unroll
        for (int j = 0; j < 2; ++j) {
            v8f cacc = {};
#pragma unroll
            for (int ks = 0; ks < 2; ++ks) {
                v16bf bk = *(const v16bf*)(&Kt[cur][(j * 16 + l15) * 64 + ks * 32 + half * 16]);
                cacc = __builtin_amdgcn_wmma_f32_16x16x32_bf16(
                    false, aq[ks], false, bk, (short)0, cacc, false, false);
            }
            s[j] = cacc;
        }

        // online softmax + stash P (bf16) into per-wave LDS
#pragma unroll
        for (int r = 0; r < 8; ++r) {
            int qg = qRow0 + r + 8 * half;                    // global q index
            float s0 = s[0][r] * 0.125f, s1 = s[1][r] * 0.125f;
            if (causal) {
                s0 = (kv0 + l15      <= qg) ? s0 : -__builtin_inff();
                s1 = (kv0 + 16 + l15 <= qg) ? s1 : -__builtin_inff();
            }
            float mx = fmaxf(s0, s1);
            mx = fmaxf(mx, __shfl_xor(mx, 1, 16));
            mx = fmaxf(mx, __shfl_xor(mx, 2, 16));
            mx = fmaxf(mx, __shfl_xor(mx, 4, 16));
            mx = fmaxf(mx, __shfl_xor(mx, 8, 16));
            float mnew = fmaxf(mrow[r], mx);
            float sc = __expf(mrow[r] - mnew);
            float p0 = __expf(s0 - mnew);
            float p1 = __expf(s1 - mnew);
            float sum = p0 + p1;
            sum += __shfl_xor(sum, 1, 16);
            sum += __shfl_xor(sum, 2, 16);
            sum += __shfl_xor(sum, 4, 16);
            sum += __shfl_xor(sum, 8, 16);
            lrow[r] = lrow[r] * sc + sum;
            mrow[r] = mnew;
#pragma unroll
            for (int nt = 0; nt < 4; ++nt) accO[nt][r] *= sc;
            int mloc = r + 8 * half;
            Pt[wave][mloc * 32 + l15]      = f2bf(p0);
            Pt[wave][mloc * 32 + 16 + l15] = f2bf(p1);
        }
        asm volatile("s_wait_dscnt 0x0" ::: "memory");   // wave-local P visible to our LDS reads

        // O += P V  (P: 16x32 A-operand from LDS; V^T in LDS is B-operand)
        v16bf ap = load_afrag(&Pt[wave][l15 * 32], half);
#pragma unroll
        for (int nt = 0; nt < 4; ++nt) {
            v16bf bv = *(const v16bf*)(&Vt[cur][(nt * 16 + l15) * 32 + half * 16]);
            accO[nt] = __builtin_amdgcn_wmma_f32_16x16x32_bf16(
                false, ap, false, bv, (short)0, accO[nt], false, false);
        }

        if (more) commitV(cur ^ 1);          // V loadcnt stall lands after the math
        wait_async0();
        __syncthreads();
    }

    // epilogue: normalize and write bf16
#pragma unroll
    for (int nt = 0; nt < 4; ++nt) {
        int col = headOff + nt * 16 + l15;
#pragma unroll
        for (int r = 0; r < 8; ++r) {
            int row = qRow0 + r + 8 * half;
            Ob[(qRowBase + row) * D + col] = f2bf(accO[nt][r] / lrow[r]);
        }
    }
}

// ---------- orchestration ----------
extern "C" void kernel_launch(void* const* d_in, const int* in_sizes, int n_in,
                              void* d_out, int out_size, void* d_ws, size_t ws_size,
                              hipStream_t stream) {
    (void)in_sizes; (void)n_in; (void)out_size; (void)ws_size;
    const int T = 1024, D = 1024, DFF = 4096, M = 4096, Tk = 1024;

    const float* dec   = (const float*)d_in[0];
    const float* enc   = (const float*)d_in[1];
    const float* sa_g  = (const float*)d_in[4];
    const float* sa_b  = (const float*)d_in[5];
    const float* sa_wq = (const float*)d_in[6];
    const float* sa_bq = (const float*)d_in[7];
    const float* sa_wk = (const float*)d_in[8];
    const float* sa_wv = (const float*)d_in[9];
    const float* sa_bv = (const float*)d_in[10];
    const float* sa_wo = (const float*)d_in[11];
    const float* sa_bo = (const float*)d_in[12];
    const float* ca_g  = (const float*)d_in[13];
    const float* ca_b  = (const float*)d_in[14];
    const float* ca_wq = (const float*)d_in[15];
    const float* ca_bq = (const float*)d_in[16];
    const float* ca_wk = (const float*)d_in[17];
    const float* ca_wv = (const float*)d_in[18];
    const float* ca_bv = (const float*)d_in[19];
    const float* ca_wo = (const float*)d_in[20];
    const float* ca_bo = (const float*)d_in[21];
    const float* ml_g  = (const float*)d_in[22];
    const float* ml_b  = (const float*)d_in[23];
    const float* w1    = (const float*)d_in[24];
    const float* b1    = (const float*)d_in[25];
    const float* w2    = (const float*)d_in[26];
    const float* b2    = (const float*)d_in[27];

    char* ws = (char*)d_ws;
    size_t off = 0;
    auto alloc = [&](size_t bytes) -> void* {
        void* p = ws + off;
        off = (off + bytes + 255) & ~(size_t)255;
        return p;
    };

    typedef unsigned short u16;
    u16* bsa_wq = (u16*)alloc((size_t)D * D * 2);
    u16* bsa_wk = (u16*)alloc((size_t)D * D * 2);
    u16* bsa_wv = (u16*)alloc((size_t)D * D * 2);
    u16* bsa_wo = (u16*)alloc((size_t)D * D * 2);
    u16* bca_wq = (u16*)alloc((size_t)D * D * 2);
    u16* bca_wk = (u16*)alloc((size_t)D * D * 2);
    u16* bca_wv = (u16*)alloc((size_t)D * D * 2);
    u16* bca_wo = (u16*)alloc((size_t)D * D * 2);
    u16* bw1    = (u16*)alloc((size_t)D * DFF * 2);
    u16* bw2    = (u16*)alloc((size_t)DFF * D * 2);
    u16* benc   = (u16*)alloc((size_t)M * D * 2);
    u16* hbuf   = (u16*)alloc((size_t)M * D * 2);
    u16* qb     = (u16*)alloc((size_t)M * D * 2);
    u16* kb     = (u16*)alloc((size_t)M * D * 2);
    u16* vb     = (u16*)alloc((size_t)M * D * 2);
    u16* abuf   = (u16*)alloc((size_t)M * D * 2);
    float* x1   = (float*)alloc((size_t)M * D * 4);
    float* x2   = (float*)alloc((size_t)M * D * 4);
    u16* h1     = (u16*)alloc((size_t)M * DFF * 2);

    auto cvt = [&](const float* s, u16* d, int n) {
        cvt_kernel<<<dim3((n + 255) / 256), dim3(256), 0, stream>>>(s, d, n);
    };
    auto gemm = [&](const u16* A, const u16* Wm, const float* bias, const float* res,
                    float* oF, u16* oB, int Mi, int Ni, int Ki, int gelu) {
        gemm_bf16_kernel<<<dim3(Ni / 128, Mi / 128), dim3(256), 0, stream>>>(
            A, Wm, bias, res, oF, oB, Mi, Ni, Ki, gelu);
    };
    auto attn = [&](const u16* Qp, const u16* Kp, const u16* Vp, u16* Op, int causal) {
        attn_kernel<<<dim3(T / 64, 64), dim3(128), 0, stream>>>(Qp, Kp, Vp, Op, T, Tk, causal);
    };

    // weight / encoder conversion to bf16
    cvt(sa_wq, bsa_wq, D * D);  cvt(sa_wk, bsa_wk, D * D);
    cvt(sa_wv, bsa_wv, D * D);  cvt(sa_wo, bsa_wo, D * D);
    cvt(ca_wq, bca_wq, D * D);  cvt(ca_wk, bca_wk, D * D);
    cvt(ca_wv, bca_wv, D * D);  cvt(ca_wo, bca_wo, D * D);
    cvt(w1, bw1, D * DFF);      cvt(w2, bw2, DFF * D);
    cvt(enc, benc, M * D);

    // ---- self-attention block ----
    ln_kernel<<<dim3(M), dim3(256), 0, stream>>>(dec, sa_g, sa_b, hbuf);
    gemm(hbuf, bsa_wq, sa_bq, nullptr, nullptr, qb, M, D, D, 0);
    gemm(hbuf, bsa_wk, nullptr, nullptr, nullptr, kb, M, D, D, 0);
    gemm(hbuf, bsa_wv, sa_bv, nullptr, nullptr, vb, M, D, D, 0);
    attn(qb, kb, vb, abuf, /*causal=*/1);
    gemm(abuf, bsa_wo, sa_bo, dec, x1, nullptr, M, D, D, 0);

    // ---- cross-attention block ----
    ln_kernel<<<dim3(M), dim3(256), 0, stream>>>(x1, ca_g, ca_b, hbuf);
    gemm(hbuf, bca_wq, ca_bq, nullptr, nullptr, qb, M, D, D, 0);
    gemm(benc, bca_wk, nullptr, nullptr, nullptr, kb, M, D, D, 0);
    gemm(benc, bca_wv, ca_bv, nullptr, nullptr, vb, M, D, D, 0);
    attn(qb, kb, vb, abuf, /*causal=*/0);
    gemm(abuf, bca_wo, ca_bo, x1, x2, nullptr, M, D, D, 0);

    // ---- GELU FFN block ----
    ln_kernel<<<dim3(M), dim3(256), 0, stream>>>(x2, ml_g, ml_b, hbuf);
    gemm(hbuf, bw1, b1, nullptr, nullptr, h1, M, DFF, D, /*gelu=*/1);
    gemm(h1, bw2, b2, x2, (float*)d_out, nullptr, M, D, DFF, 0);
}
